// SemanticSTViT_73796128080535
// MI455X (gfx1250) — matrix operand order repaired and checked by reading, not compiled, not run on hardware
//
#include <hip/hip_runtime.h>

typedef __attribute__((ext_vector_type(16))) __bf16        v16bf;
typedef __attribute__((ext_vector_type(8)))  float         v8f;
typedef __attribute__((ext_vector_type(4)))  unsigned int  v4u;
typedef __attribute__((ext_vector_type(8)))  unsigned int  v8u;

#define WS_    7
#define N_TOK  49
#define NH_    6
#define C_     192
#define HID_   768
#define HW     56
#define M_TOK  200704          // B * nW * N = 64*64*49
#define SCALE_ 0.17677669529663689f   // 32^-0.5

static __device__ __forceinline__ unsigned short f2bf(float f) {
  unsigned u = __float_as_uint(f);
  unsigned r = ((u >> 16) & 1u) + 0x7fffu;   // round-to-nearest-even
  return (unsigned short)((u + r) >> 16);
}

static __device__ __forceinline__ v8f zero8() {
  v8f z = {0.f,0.f,0.f,0.f,0.f,0.f,0.f,0.f};
  return z;
}

// Build a 16xbf16 fragment from two 16-byte chunks (elems 0-7 from p0, 8-15 from p1).
static __device__ __forceinline__ v16bf ld_frag(const unsigned short* p0,
                                                const unsigned short* p1) {
  const v4u a = *(const v4u*)p0;
  const v4u b = *(const v4u*)p1;
  union { v8u u; v16bf v; } f;
  f.u[0]=a[0]; f.u[1]=a[1]; f.u[2]=a[2]; f.u[3]=a[3];
  f.u[4]=b[0]; f.u[5]=b[1]; f.u[6]=b[2]; f.u[7]=b[3];
  return f.v;
}

// CDNA5 async DMA: 16B per lane, global -> LDS, tracked by ASYNCcnt.
// Flat pointers to LDS carry the LDS byte offset in their low 32 bits
// (aperture rule).
static __device__ __forceinline__ void async_ld16(const void* gaddr, void* lds) {
  unsigned l = (unsigned)(uintptr_t)lds;
  asm volatile("global_load_async_to_lds_b128 %0, %1, off"
               :: "v"(l), "v"(gaddr) : "memory");
}
static __device__ __forceinline__ void wait_async0() {
  asm volatile("s_wait_asynccnt 0x0" ::: "memory");
}

// ---------------------------------------------------------------------------
// Weight convert fp32 [K,N] -> bf16 transposed [N,K]
// ---------------------------------------------------------------------------
__global__ void swin_wcvt_kernel(const float* __restrict__ w,
                                 unsigned short* __restrict__ wt,
                                 int K, int N) {
  int i = blockIdx.x * blockDim.x + threadIdx.x;
  if (i >= K * N) return;
  int k = i / N, n = i % N;
  wt[(size_t)n * K + k] = f2bf(w[i]);
}

// ---------------------------------------------------------------------------
// LayerNorm (one wave per token).  windowed=1: gather with cyclic shift +
// window partition (dst row is window-major), windowed=0: identity mapping.
// ---------------------------------------------------------------------------
__global__ __launch_bounds__(256) void swin_ln_kernel(
    const float* __restrict__ xin, const float* __restrict__ g,
    const float* __restrict__ be, unsigned short* __restrict__ outb,
    int windowed) {
  const int lane = threadIdx.x & 31;
  const int wv   = threadIdx.x >> 5;
  const int row  = blockIdx.x * 8 + wv;
  const float* src;
  if (windowed) {
    int win = row / N_TOK, t = row % N_TOK;
    int b_ = win >> 6, wrem = win & 63;
    int hs  = (wrem >> 3) * WS_ + t / WS_;
    int ws2 = (wrem & 7)  * WS_ + t % WS_;
    int hp = hs + 3;  if (hp >= HW) hp -= HW;
    int wp = ws2 + 3; if (wp >= HW) wp -= HW;
    src = xin + ((size_t)b_ * (HW*HW) + hp * HW + wp) * C_;
  } else {
    src = xin + (size_t)row * C_;
  }
  float vals[6]; float s = 0.f, s2 = 0.f;
#pragma unroll
  for (int i = 0; i < 6; ++i) {
    float v = src[lane + 32*i]; vals[i] = v; s += v; s2 += v*v;
  }
#pragma unroll
  for (int xm = 16; xm >= 1; xm >>= 1) {
    s += __shfl_xor(s, xm, 32); s2 += __shfl_xor(s2, xm, 32);
  }
  const float mu = s * (1.0f / C_);
  const float rs = rsqrtf(s2 * (1.0f / C_) - mu * mu + 1e-5f);
  unsigned short* dst = outb + (size_t)row * C_;
#pragma unroll
  for (int i = 0; i < 6; ++i) {
    int c = lane + 32*i;
    dst[c] = f2bf((vals[i] - mu) * rs * g[c] + be[c]);
  }
}

// ---------------------------------------------------------------------------
// bf16 WMMA GEMM: C[M,N] = A[M,K] * Bt[N,K]^T (+epilogue)
// BM=128, BN=64, BK=32; 256 threads = 8 waves (4 along M x 2 along N),
// each wave a 32x32 output tile (4 WMMAs / k-step).
// Double-buffered LDS filled by global_load_async_to_lds_b128 (ASYNCcnt),
// so tile k+1 streams in while tile k is multiplied.
// mode 0: +bias, store bf16            (qkv)
// mode 1: +bias, window-reverse+unshift scatter, +x residual, store f32 (proj)
// mode 2: +bias, exact GELU, store bf16 (fc1)
// mode 3: +bias, +extra residual, store f32 (fc2 -> d_out)
// ---------------------------------------------------------------------------
#define BM 128
#define BN 64
#define BK 32
#define LDT 40   // padded LDS row stride (ushorts): 80B = 20 banks -> conflict-free

__global__ __launch_bounds__(256) void swin_gemm_kernel(
    const unsigned short* __restrict__ A,
    const unsigned short* __restrict__ Bt,
    const float* __restrict__ bias,
    const float* __restrict__ extra,
    void* __restrict__ outp,
    int N, int K, int mode) {
  __shared__ __align__(16) unsigned short As[2][BM * LDT];
  __shared__ __align__(16) unsigned short Bs[2][BN * LDT];
  const int tid  = threadIdx.x;
  const int lane = tid & 31;
  const int wv   = tid >> 5;
  const int wm   = (wv & 3) << 5;   // wave M offset in block tile (0,32,64,96)
  const int wn   = (wv >> 2) << 5;  // wave N offset (0,32)
  const size_t bm = (size_t)blockIdx.y * BM;
  const int    bn = blockIdx.x * BN;
  const int lr  = tid >> 2;         // coop row 0..63
  const int lc  = (tid & 3) << 3;   // coop col chunk {0,8,16,24}
  const int nl  = lane & 15;
  const int kbA = (lane < 16) ? 0 : 16;   // A-fragment lane-half K base (contiguous k = 16h + s)
  const int kbB = (lane < 16) ? 0 : 16;   // B-fragment lane-half K base (contiguous k = 16h + s)
  const int nk  = K / BK;

  v8f acc00 = zero8(), acc01 = zero8(), acc10 = zero8(), acc11 = zero8();

  auto issue_tiles = [&](int k0, int bufi) {
    async_ld16(A  + (bm + lr)      * K + k0 + lc, &As[bufi][lr        * LDT + lc]);
    async_ld16(A  + (bm + 64 + lr) * K + k0 + lc, &As[bufi][(64 + lr) * LDT + lc]);
    async_ld16(Bt + (size_t)(bn + lr) * K + k0 + lc, &Bs[bufi][lr * LDT + lc]);
  };

  issue_tiles(0, 0);
  for (int ks = 0; ks < nk; ++ks) {
    const int cur = ks & 1;
    wait_async0();          // this wave's copies for tile ks complete
    __syncthreads();        // all waves' copies complete; prior reads of the
                            // spare buffer finished (dscnt waited before wmma)
    if (ks + 1 < nk) issue_tiles((ks + 1) * BK, cur ^ 1);

    const unsigned short* a0 = &As[cur][(wm + nl)      * LDT];
    const unsigned short* a1 = &As[cur][(wm + 16 + nl) * LDT];
    const unsigned short* b0 = &Bs[cur][(wn + nl)      * LDT + kbB];
    const unsigned short* b1 = &Bs[cur][(wn + 16 + nl) * LDT + kbB];
    v16bf af0 = ld_frag(a0 + kbA, a0 + 8 + kbA);
    v16bf af1 = ld_frag(a1 + kbA, a1 + 8 + kbA);
    v16bf bf0 = ld_frag(b0, b0 + 8);
    v16bf bf1 = ld_frag(b1, b1 + 8);
    acc00 = __builtin_amdgcn_wmma_f32_16x16x32_bf16(false, af0, false, bf0,
                                                    (short)0, acc00, false, false);
    acc01 = __builtin_amdgcn_wmma_f32_16x16x32_bf16(false, af0, false, bf1,
                                                    (short)0, acc01, false, false);
    acc10 = __builtin_amdgcn_wmma_f32_16x16x32_bf16(false, af1, false, bf0,
                                                    (short)0, acc10, false, false);
    acc11 = __builtin_amdgcn_wmma_f32_16x16x32_bf16(false, af1, false, bf1,
                                                    (short)0, acc11, false, false);
    __syncthreads();        // done reading buf[cur] before it is refilled
  }

  union { v8f v; float f[8]; } u[2][2];
  u[0][0].v = acc00; u[0][1].v = acc01; u[1][0].v = acc10; u[1][1].v = acc11;
  const int mgrp = (lane >> 4) << 3;
#pragma unroll
  for (int rt = 0; rt < 2; ++rt) {
#pragma unroll
    for (int rr = 0; rr < 8; ++rr) {
      size_t m = bm + wm + rt * 16 + rr + mgrp;
#pragma unroll
      for (int nt = 0; nt < 2; ++nt) {
        int n = bn + wn + (nt << 4) + nl;
        float v = u[rt][nt].f[rr] + bias[n];
        if (mode == 0) {
          ((unsigned short*)outp)[m * N + n] = f2bf(v);
        } else if (mode == 1) {
          int mi = (int)m;
          int win = mi / N_TOK, t = mi % N_TOK;
          int b_ = win >> 6, wrem = win & 63;
          int hs  = (wrem >> 3) * WS_ + t / WS_;
          int ws2 = (wrem & 7)  * WS_ + t % WS_;
          int hp = hs + 3;  if (hp >= HW) hp -= HW;
          int wp = ws2 + 3; if (wp >= HW) wp -= HW;
          size_t idx = ((size_t)b_ * (HW*HW) + hp * HW + wp) * C_ + n;
          ((float*)outp)[idx] = extra[idx] + v;
        } else if (mode == 2) {
          float gch = 0.5f * v * (1.0f + erff(v * 0.70710678118654752f));
          ((unsigned short*)outp)[m * N + n] = f2bf(gch);
        } else {
          size_t idx = m * N + n;
          ((float*)outp)[idx] = extra[idx] + v;
        }
      }
    }
  }
}

// ---------------------------------------------------------------------------
// Windowed attention: one block (128 thr = 4 waves) per (window, head).
// Tokens padded 49->64 in LDS. QK^T, rel-pos bias (index recomputed), shift
// mask, softmax (half-wave shuffles), AV.  All matmuls are bf16 WMMA.
// ---------------------------------------------------------------------------
__global__ __launch_bounds__(128) void swin_attn_kernel(
    const unsigned short* __restrict__ qkv,   // [M_TOK, 576] bf16
    const float* __restrict__ rpb,            // [169, 6]
    const float* __restrict__ maskm,          // [64, 49, 49]
    unsigned short* __restrict__ aout) {      // [M_TOK, 192] bf16
  __shared__ __align__(16) unsigned short Qs[64 * 32];
  __shared__ __align__(16) unsigned short Ks[64 * 32];
  __shared__ __align__(16) unsigned short Vt[32 * 64];  // transposed: [d][token]
  __shared__ __align__(16) unsigned short Ps[64 * 64];  // softmax probs (bf16)

  const int tid  = threadIdx.x;
  const int lane = tid & 31;
  const int wv   = tid >> 5;
  const int blk  = blockIdx.x;
  const int win  = blk / NH_;
  const int h    = blk % NH_;
  const int wimg = win & 63;
  const size_t rowbase = (size_t)win * N_TOK;

  { // stage Q (threads 0-63) and K (threads 64-127), zero-padded to 64 rows
    int t = tid & 63;
    int which = tid >> 6;   // 0 -> Q, 1 -> K
    v4u d0 = 0, d1 = 0, d2 = 0, d3 = 0;
    if (t < N_TOK) {
      const v4u* s4 = (const v4u*)(qkv + (rowbase + t) * 576 + which * 192 + h * 32);
      d0 = s4[0]; d1 = s4[1]; d2 = s4[2]; d3 = s4[3];
    }
    v4u* dst = (v4u*)((which ? Ks : Qs) + t * 32);
    dst[0] = d0; dst[1] = d1; dst[2] = d2; dst[3] = d3;
  }
  { // stage V transposed (threads 0-63)
    int t = tid & 63;
    if ((tid >> 6) == 0) {
      if (t < N_TOK) {
        const unsigned short* src = qkv + (rowbase + t) * 576 + 384 + h * 32;
#pragma unroll
        for (int d = 0; d < 32; ++d) Vt[d * 64 + t] = src[d];
      } else {
#pragma unroll
        for (int d = 0; d < 32; ++d) Vt[d * 64 + t] = 0;
      }
    }
  }
  __syncthreads();

  const int m0  = wv * 16;
  const int nl  = lane & 15;
  const int kbA = (lane < 16) ? 0 : 16;
  const int kbB = (lane < 16) ? 0 : 16;

  // ---- scores = Q * K^T   (4 column tiles of 16) ----
  const unsigned short* qrow = Qs + (m0 + nl) * 32;
  v16bf qf = ld_frag(qrow + kbA, qrow + 8 + kbA);
  union { v8f v; float f[8]; } sc[4];
#pragma unroll
  for (int nt = 0; nt < 4; ++nt) {
    const unsigned short* kr = Ks + (nt * 16 + nl) * 32 + kbB;
    v16bf kf = ld_frag(kr, kr + 8);
    sc[nt].v = __builtin_amdgcn_wmma_f32_16x16x32_bf16(false, qf, false, kf,
                                                       (short)0, zero8(), false, false);
  }

  // ---- bias + mask + softmax (rows live in 16-lane halves) ----
  const int mgrp = lane >> 4;
#pragma unroll
  for (int rr = 0; rr < 8; ++rr) {
    int m  = m0 + rr + 8 * mgrp;
    int mc = (m < N_TOK) ? m : (N_TOK - 1);
    int mi = mc / WS_, mj = mc % WS_;
    float vals[4];
    float mx = -1e30f;
#pragma unroll
    for (int nt = 0; nt < 4; ++nt) {
      int n = nt * 16 + nl;
      float v;
      if (n < N_TOK) {
        int ni = n / WS_, nj = n % WS_;
        int ridx = (mi - ni + (WS_ - 1)) * (2 * WS_ - 1) + (mj - nj + (WS_ - 1));
        v = sc[nt].f[rr] * SCALE_ + rpb[ridx * NH_ + h]
            + maskm[(size_t)wimg * (N_TOK * N_TOK) + mc * N_TOK + n];
      } else {
        v = -1e30f;
      }
      vals[nt] = v;
      mx = fmaxf(mx, v);
    }
#pragma unroll
    for (int xm = 8; xm >= 1; xm >>= 1) mx = fmaxf(mx, __shfl_xor(mx, xm, 32));
    float s = 0.f;
#pragma unroll
    for (int nt = 0; nt < 4; ++nt) {
      float e = __expf(vals[nt] - mx); vals[nt] = e; s += e;
    }
#pragma unroll
    for (int xm = 8; xm >= 1; xm >>= 1) s += __shfl_xor(s, xm, 32);
    float inv = 1.0f / s;
#pragma unroll
    for (int nt = 0; nt < 4; ++nt)
      Ps[m * 64 + nt * 16 + nl] = f2bf(vals[nt] * inv);
  }
  __syncthreads();

  // ---- out = P * V  (2 d-tiles of 16, K loop 64 in 2 steps) ----
  v8f oacc0 = zero8(), oacc1 = zero8();
#pragma unroll
  for (int ks = 0; ks < 2; ++ks) {
    const unsigned short* prow = Ps + (m0 + nl) * 64 + 32 * ks;
    v16bf pf = ld_frag(prow + kbA, prow + 8 + kbA);
    const unsigned short* vr0 = Vt + (nl) * 64 + 32 * ks + kbB;
    const unsigned short* vr1 = Vt + (16 + nl) * 64 + 32 * ks + kbB;
    v16bf vf0 = ld_frag(vr0, vr0 + 8);
    v16bf vf1 = ld_frag(vr1, vr1 + 8);
    oacc0 = __builtin_amdgcn_wmma_f32_16x16x32_bf16(false, pf, false, vf0,
                                                    (short)0, oacc0, false, false);
    oacc1 = __builtin_amdgcn_wmma_f32_16x16x32_bf16(false, pf, false, vf1,
                                                    (short)0, oacc1, false, false);
  }
  union { v8f v; float f[8]; } o0, o1;
  o0.v = oacc0; o1.v = oacc1;
#pragma unroll
  for (int rr = 0; rr < 8; ++rr) {
    int m = m0 + rr + 8 * mgrp;
    if (m < N_TOK) {
      size_t ro = (rowbase + m) * C_ + h * 32;
      aout[ro + nl]      = f2bf(o0.f[rr]);
      aout[ro + 16 + nl] = f2bf(o1.f[rr]);
    }
  }
}

// ---------------------------------------------------------------------------
extern "C" void kernel_launch(void* const* d_in, const int* in_sizes, int n_in,
                              void* d_out, int out_size, void* d_ws, size_t ws_size,
                              hipStream_t stream) {
  (void)in_sizes; (void)n_in; (void)out_size; (void)ws_size;
  const float* x      = (const float*)d_in[0];
  const float* maskm  = (const float*)d_in[1];
  const float* n1g    = (const float*)d_in[2];
  const float* n1b    = (const float*)d_in[3];
  const float* qkv_w  = (const float*)d_in[4];
  const float* qkv_b  = (const float*)d_in[5];
  const float* rpb    = (const float*)d_in[6];
  const float* proj_w = (const float*)d_in[7];
  const float* proj_b = (const float*)d_in[8];
  const float* n2g    = (const float*)d_in[9];
  const float* n2b    = (const float*)d_in[10];
  const float* fc1_w  = (const float*)d_in[11];
  const float* fc1_b  = (const float*)d_in[12];
  const float* fc2_w  = (const float*)d_in[13];
  const float* fc2_b  = (const float*)d_in[14];

  const int M = M_TOK;
  char* p = (char*)d_ws;
  auto carve = [&](size_t bytes) {
    char* q = p; p += (bytes + 255) & ~(size_t)255; return q;
  };
  // buf1: xw -> attn_out -> h2   (bf16 [M,192])
  unsigned short* buf1 = (unsigned short*)carve((size_t)M * C_ * 2);
  // buf2: qkv (bf16 [M,576]) -> hidden (bf16 [M,768])
  unsigned short* buf2 = (unsigned short*)carve((size_t)M * HID_ * 2);
  // buf3: y = x + attn residual (f32 [M,192], image order)
  float* yb = (float*)carve((size_t)M * C_ * 4);
  unsigned short* qkvWt  = (unsigned short*)carve((size_t)576 * C_ * 2);
  unsigned short* projWt = (unsigned short*)carve((size_t)C_ * C_ * 2);
  unsigned short* fc1Wt  = (unsigned short*)carve((size_t)HID_ * C_ * 2);
  unsigned short* fc2Wt  = (unsigned short*)carve((size_t)C_ * HID_ * 2);

  // 1) weights -> bf16 transposed [N,K]
  swin_wcvt_kernel<<<(C_*576 + 255)/256, 256, 0, stream>>>(qkv_w, qkvWt, C_, 576);
  swin_wcvt_kernel<<<(C_*C_  + 255)/256, 256, 0, stream>>>(proj_w, projWt, C_, C_);
  swin_wcvt_kernel<<<(C_*HID_+ 255)/256, 256, 0, stream>>>(fc1_w, fc1Wt, C_, HID_);
  swin_wcvt_kernel<<<(HID_*C_+ 255)/256, 256, 0, stream>>>(fc2_w, fc2Wt, HID_, C_);

  // 2) LN1 + cyclic shift + window partition -> buf1 (windowed bf16)
  swin_ln_kernel<<<M/8, 256, 0, stream>>>(x, n1g, n1b, buf1, 1);

  // 3) QKV GEMM: [M,192] x [192,576] -> buf2
  swin_gemm_kernel<<<dim3(576/BN, M/BM), 256, 0, stream>>>(
      buf1, qkvWt, qkv_b, nullptr, buf2, 576, C_, 0);

  // 4) windowed attention -> buf1 (attn_out, windowed bf16)
  swin_attn_kernel<<<(M / N_TOK) * NH_, 128, 0, stream>>>(buf2, rpb, maskm, buf1);

  // 5) proj GEMM + window reverse + unshift + residual -> yb (f32, image order)
  swin_gemm_kernel<<<dim3(C_/BN, M/BM), 256, 0, stream>>>(
      buf1, projWt, proj_b, x, yb, C_, C_, 1);

  // 6) LN2 -> buf1 (h2 bf16, image order)
  swin_ln_kernel<<<M/8, 256, 0, stream>>>(yb, n2g, n2b, buf1, 0);

  // 7) fc1 GEMM + GELU -> buf2 (hidden bf16)
  swin_gemm_kernel<<<dim3(HID_/BN, M/BM), 256, 0, stream>>>(
      buf1, fc1Wt, fc1_b, nullptr, buf2, HID_, C_, 2);

  // 8) fc2 GEMM + residual -> d_out (f32)
  swin_gemm_kernel<<<dim3(C_/BN, M/BM), 256, 0, stream>>>(
      buf2, fc2Wt, fc2_b, yb, d_out, C_, HID_, 3);
}
